// attentionblock_37503654429022
// MI455X (gfx1250) — compile-verified, matrix operand order checked by reading, not run on hardware
//
#include <hip/hip_runtime.h>

typedef __attribute__((ext_vector_type(16))) __bf16 v16bf;
typedef __attribute__((ext_vector_type(8)))  __bf16 v8bf;
typedef __attribute__((ext_vector_type(8)))  float  v8f;

__device__ __forceinline__ __bf16 f2bf(float f) {
    unsigned int u = __builtin_bit_cast(unsigned int, f);
    unsigned int r = (u + 0x7FFFu + ((u >> 16) & 1u)) >> 16;
    unsigned short s = (unsigned short)r;
    return __builtin_bit_cast(__bf16, s);
}

__device__ __forceinline__ v8f zero8() {
    v8f z = {0.f,0.f,0.f,0.f,0.f,0.f,0.f,0.f};
    return z;
}

// Load a 16-element bf16 WMMA fragment: elements 0..7 = p[0..7], 8..15 = p[16..23]
__device__ __forceinline__ v16bf ld_frag(const __bf16* p) {
    union { v16bf v; v8bf h[2]; } u;
    u.h[0] = *(const v8bf*)(p);
    u.h[1] = *(const v8bf*)(p + 16);
    return u.v;
}

// ---------------------------------------------------------------------------
// Conversion / packing kernels
// ---------------------------------------------------------------------------
__global__ void cvt_bf16_kernel(const float* __restrict__ in,
                                __bf16* __restrict__ out, int n) {
    for (int i = blockIdx.x * blockDim.x + threadIdx.x; i < n;
         i += gridDim.x * blockDim.x)
        out[i] = f2bf(in[i]);
}

// WqkvT[n][c], n in [0,3072): n<1024 -> Q col (h*64+d), then K, then V.
// Attention scale 1/sqrt(C) = 1/32 is folded into the Q weights + bias.
__global__ void pack_qkv_kernel(const float* __restrict__ Wq,
                                const float* __restrict__ Wk,
                                const float* __restrict__ Wv,
                                const float* __restrict__ bq,
                                const float* __restrict__ bk,
                                const float* __restrict__ bv,
                                __bf16* __restrict__ Wt,
                                float* __restrict__ bias) {
    const int total = 3072 * 1024;
    for (int idx = blockIdx.x * blockDim.x + threadIdx.x; idx < total;
         idx += gridDim.x * blockDim.x) {
        int n = idx >> 10, c = idx & 1023;
        int sel = n >> 10, r = n & 1023;
        int h = r >> 6, d = r & 63;
        const float* W = (sel == 0) ? Wq : (sel == 1) ? Wk : Wv;
        float w = W[((size_t)h * 1024 + c) * 64 + d];
        if (sel == 0) w *= 0.03125f;
        Wt[idx] = f2bf(w);
    }
    for (int n = blockIdx.x * blockDim.x + threadIdx.x; n < 3072;
         n += gridDim.x * blockDim.x) {
        int sel = n >> 10, r = n & 1023;
        const float* bb = (sel == 0) ? bq : (sel == 1) ? bk : bv;
        float w = bb[r];
        if (sel == 0) w *= 0.03125f;
        bias[n] = w;
    }
}

// out[n*K + k] = bf16(in[k*N + n])
__global__ void transp_bf16_kernel(const float* __restrict__ in,
                                   __bf16* __restrict__ out, int N, int K) {
    const int total = N * K;
    for (int idx = blockIdx.x * blockDim.x + threadIdx.x; idx < total;
         idx += gridDim.x * blockDim.x) {
        int n = idx / K, k = idx % K;
        out[idx] = f2bf(in[(size_t)k * N + n]);
    }
}

// ---------------------------------------------------------------------------
// Generic bf16 WMMA GEMM:  C[M][N] = A[M][K] * Bt[N][K]^T  (+bias, epilogue)
// EPI: 0 = bf16 out (+bias), 1 = bf16 out (+bias, relu), 2 = f32 out (+bias)
// Double-buffered LDS tiles; bulk fragment loads; 16 WMMAs per K-tile/wave.
// ---------------------------------------------------------------------------
template<int EPI>
__global__ __launch_bounds__(256) void gemm_bf16_kernel(
    const __bf16* __restrict__ A, const __bf16* __restrict__ Bt,
    const float* __restrict__ bias, void* __restrict__ Cout,
    int M, int N, int K) {
    constexpr int BM = 128, BN = 128, BK = 64, LDT = 72;
    __shared__ __bf16 As[2][BM * LDT];
    __shared__ __bf16 Bs[2][BN * LDT];

    const int tid  = threadIdx.x;
    const int lane = tid & 31, wave = tid >> 5;
    const int wm = wave & 3, wn = wave >> 2;     // 4 waves on M, 2 on N
    const int lrow = lane & 15, lhi = lane >> 4;
    const int m0 = blockIdx.y * BM, n0 = blockIdx.x * BN;

    v8f acc[2][4];
#pragma unroll
    for (int mi = 0; mi < 2; ++mi)
#pragma unroll
        for (int ni = 0; ni < 4; ++ni) acc[mi][ni] = zero8();

    auto load_tile = [&](int k0, int buf) {
#pragma unroll
        for (int i = 0; i < 4; ++i) {
            int vid = tid + i * 256;
            int r = vid >> 3, cv = (vid & 7) * 8;
            *(v8bf*)(As[buf] + r * LDT + cv) =
                *(const v8bf*)(A + (size_t)(m0 + r) * K + k0 + cv);
            *(v8bf*)(Bs[buf] + r * LDT + cv) =
                *(const v8bf*)(Bt + (size_t)(n0 + r) * K + k0 + cv);
        }
    };

    const int nk = K / BK;
    load_tile(0, 0);
    int cur = 0;
    for (int kt = 0; kt < nk; ++kt) {
        __syncthreads();
        if (kt + 1 < nk) load_tile((kt + 1) * BK, cur ^ 1);

        v16bf af[2][2], bfm[2][4];
#pragma unroll
        for (int ks = 0; ks < 2; ++ks) {
            const int kb = ks * 32 + lhi * 8;
#pragma unroll
            for (int mi = 0; mi < 2; ++mi)
                af[ks][mi] =
                    ld_frag(As[cur] + (wm * 32 + mi * 16 + lrow) * LDT + kb);
#pragma unroll
            for (int ni = 0; ni < 4; ++ni)
                bfm[ks][ni] =
                    ld_frag(Bs[cur] + (wn * 64 + ni * 16 + lrow) * LDT + kb);
        }
#pragma unroll
        for (int ks = 0; ks < 2; ++ks)
#pragma unroll
            for (int mi = 0; mi < 2; ++mi)
#pragma unroll
                for (int ni = 0; ni < 4; ++ni)
                    acc[mi][ni] = __builtin_amdgcn_wmma_f32_16x16x32_bf16(
                        false, af[ks][mi], false, bfm[ks][ni], (short)0,
                        acc[mi][ni], false, false);
        cur ^= 1;
    }

#pragma unroll
    for (int mi = 0; mi < 2; ++mi)
#pragma unroll
        for (int ni = 0; ni < 4; ++ni)
#pragma unroll
            for (int r = 0; r < 8; ++r) {
                int m = m0 + wm * 32 + mi * 16 + r + 8 * lhi;
                int n = n0 + wn * 64 + ni * 16 + lrow;
                float v = acc[mi][ni][r] + bias[n];
                if (EPI == 2) {
                    ((float*)Cout)[(size_t)m * N + n] = v;
                } else {
                    if (EPI == 1) v = fmaxf(v, 0.f);
                    ((__bf16*)Cout)[(size_t)m * N + n] = f2bf(v);
                }
            }
}

// ---------------------------------------------------------------------------
// Flash attention: qkv is [8192][3072] bf16 (q|k|v, head-major cols).
// One WG = one (b,h) and 64 q-rows; 4 waves, each wave owns 16 q-rows.
// Q is pre-scaled by 1/sqrt(C) at weight-pack time. K/V tiles double-buffered.
// ---------------------------------------------------------------------------
__global__ __launch_bounds__(128) void attn_kernel(
    const __bf16* __restrict__ qkv, __bf16* __restrict__ attnb) {
    constexpr int LD = 72;
    __shared__ __bf16 kt[2][64 * LD];
    __shared__ __bf16 vt[2][64 * LD];
    __shared__ __bf16 pt[64 * LD];

    const int tid = threadIdx.x, lane = tid & 31, wave = tid >> 5;
    const int lrow = lane & 15, lhi = lane >> 4;
    const int qb = blockIdx.x;           // 0..15
    const int bh = blockIdx.y;           // 0..127
    const int b = bh >> 4, h = bh & 15;
    const size_t rowbase = (size_t)b * 1024 * 3072;
    const __bf16* qp = qkv + rowbase + h * 64;
    const __bf16* kp = qkv + rowbase + 1024 + h * 64;
    const __bf16* vp = qkv + rowbase + 2048 + h * 64;
    const int q0 = qb * 64;

    // Q fragments for this wave's 16 rows (K = D = 64 -> 2 k-steps), kept live.
    v16bf aq[2];
    {
        const size_t mr = (size_t)(q0 + wave * 16 + lrow) * 3072;
#pragma unroll
        for (int ks = 0; ks < 2; ++ks) {
            int kb = ks * 32 + lhi * 8;
            union { v16bf v; v8bf h2[2]; } u;
            u.h2[0] = *(const v8bf*)(qp + mr + kb);
            u.h2[1] = *(const v8bf*)(qp + mr + kb + 16);
            aq[ks] = u.v;
        }
    }

    v8f oacc[4];
#pragma unroll
    for (int ni = 0; ni < 4; ++ni) oacc[ni] = zero8();
    float mrow[8], lsum[8];
#pragma unroll
    for (int r = 0; r < 8; ++r) { mrow[r] = -1e30f; lsum[r] = 0.f; }

    auto load_kv = [&](int s0, int buf) {
#pragma unroll
        for (int i = 0; i < 4; ++i) {
            int vid = tid + i * 128;
            int r = vid >> 3, cv = (vid & 7) * 8;
            *(v8bf*)(kt[buf] + r * LD + cv) =
                *(const v8bf*)(kp + (size_t)(s0 + r) * 3072 + cv);
            v8bf vv = *(const v8bf*)(vp + (size_t)(s0 + r) * 3072 + cv);
#pragma unroll
            for (int j = 0; j < 8; ++j) vt[buf][(cv + j) * LD + r] = vv[j];
        }
    };

    load_kv(0, 0);
    int cur = 0;
    for (int it = 0; it < 16; ++it) {
        __syncthreads();
        if (it + 1 < 16) load_kv((it + 1) * 64, cur ^ 1);

        // S = Q * K^T  (16x64 per wave)
        v8f sacc[4];
#pragma unroll
        for (int ni = 0; ni < 4; ++ni) sacc[ni] = zero8();
#pragma unroll
        for (int ks = 0; ks < 2; ++ks) {
            const int kb = ks * 32 + lhi * 8;
            v16bf bk16[4];
#pragma unroll
            for (int ni = 0; ni < 4; ++ni)
                bk16[ni] = ld_frag(kt[cur] + (ni * 16 + lrow) * LD + kb);
#pragma unroll
            for (int ni = 0; ni < 4; ++ni)
                sacc[ni] = __builtin_amdgcn_wmma_f32_16x16x32_bf16(
                    false, aq[ks], false, bk16[ni], (short)0, sacc[ni], false,
                    false);
        }

        // Online softmax per row (rows r+8*lhi of this wave's 16).
#pragma unroll
        for (int r = 0; r < 8; ++r) {
            float tm = -1e30f;
#pragma unroll
            for (int ni = 0; ni < 4; ++ni) tm = fmaxf(tm, sacc[ni][r]);
#pragma unroll
            for (int mk = 1; mk < 16; mk <<= 1)
                tm = fmaxf(tm, __shfl_xor(tm, mk, 32));
            float mn = fmaxf(mrow[r], tm);
            float corr = __expf(mrow[r] - mn);
            float rs = 0.f;
            const int prow = (wave * 16 + r + 8 * lhi) * LD;
#pragma unroll
            for (int ni = 0; ni < 4; ++ni) {
                float p = __expf(sacc[ni][r] - mn);
                rs += p;
                pt[prow + ni * 16 + lrow] = f2bf(p);
            }
#pragma unroll
            for (int mk = 1; mk < 16; mk <<= 1) rs += __shfl_xor(rs, mk, 32);
            lsum[r] = lsum[r] * corr + rs;
#pragma unroll
            for (int ni = 0; ni < 4; ++ni) oacc[ni][r] *= corr;
            mrow[r] = mn;
        }

        // O += P * V
#pragma unroll
        for (int ks = 0; ks < 2; ++ks) {
            const int kb = ks * 32 + lhi * 8;
            v16bf ap = ld_frag(pt + (wave * 16 + lrow) * LD + kb);
            v16bf bv16[4];
#pragma unroll
            for (int ni = 0; ni < 4; ++ni)
                bv16[ni] = ld_frag(vt[cur] + (ni * 16 + lrow) * LD + kb);
#pragma unroll
            for (int ni = 0; ni < 4; ++ni)
                oacc[ni] = __builtin_amdgcn_wmma_f32_16x16x32_bf16(
                    false, ap, false, bv16[ni], (short)0, oacc[ni], false,
                    false);
        }
        cur ^= 1;
    }

#pragma unroll
    for (int ni = 0; ni < 4; ++ni)
#pragma unroll
        for (int r = 0; r < 8; ++r) {
            int t = q0 + wave * 16 + r + 8 * lhi;
            float o = oacc[ni][r] / lsum[r];
            attnb[(size_t)(b * 1024 + t) * 1024 + h * 64 + ni * 16 + lrow] =
                f2bf(o);
        }
}

// ---------------------------------------------------------------------------
// out[row] = LayerNorm(a[row] + b[row]) * g + be   (C = 1024, eps = 1e-5)
// ---------------------------------------------------------------------------
__global__ __launch_bounds__(256) void ln_add_kernel(
    const float* __restrict__ a, const float* __restrict__ bsrc,
    const float* __restrict__ g, const float* __restrict__ be,
    float* __restrict__ out) {
    __shared__ float red[256];
    const int row = blockIdx.x, tid = threadIdx.x;
    const size_t base = (size_t)row * 1024;
    float v[4];
    float s = 0.f;
#pragma unroll
    for (int i = 0; i < 4; ++i) {
        int c = tid + i * 256;
        v[i] = a[base + c] + bsrc[base + c];
        s += v[i];
    }
    red[tid] = s; __syncthreads();
    for (int off = 128; off > 0; off >>= 1) {
        if (tid < off) red[tid] += red[tid + off];
        __syncthreads();
    }
    const float mu = red[0] * (1.0f / 1024.0f);
    __syncthreads();
    s = 0.f;
#pragma unroll
    for (int i = 0; i < 4; ++i) { float d = v[i] - mu; s += d * d; }
    red[tid] = s; __syncthreads();
    for (int off = 128; off > 0; off >>= 1) {
        if (tid < off) red[tid] += red[tid + off];
        __syncthreads();
    }
    const float rstd = rsqrtf(red[0] * (1.0f / 1024.0f) + 1e-5f);
#pragma unroll
    for (int i = 0; i < 4; ++i) {
        int c = tid + i * 256;
        out[base + c] = (v[i] - mu) * rstd * g[c] + be[c];
    }
}

// ---------------------------------------------------------------------------
extern "C" void kernel_launch(void* const* d_in, const int* in_sizes, int n_in,
                              void* d_out, int out_size, void* d_ws,
                              size_t ws_size, hipStream_t stream) {
    (void)in_sizes; (void)n_in; (void)out_size; (void)ws_size;
    const float* x   = (const float*)d_in[0];
    const float* Wq  = (const float*)d_in[1];
    const float* bq  = (const float*)d_in[2];
    const float* Wk  = (const float*)d_in[3];
    const float* bk  = (const float*)d_in[4];
    const float* Wv  = (const float*)d_in[5];
    const float* bv  = (const float*)d_in[6];
    const float* Wo  = (const float*)d_in[7];
    const float* bo  = (const float*)d_in[8];
    const float* W1  = (const float*)d_in[9];
    const float* b1  = (const float*)d_in[10];
    const float* W2  = (const float*)d_in[11];
    const float* b2  = (const float*)d_in[12];
    const float* g1  = (const float*)d_in[13];
    const float* be1 = (const float*)d_in[14];
    const float* g2  = (const float*)d_in[15];
    const float* be2 = (const float*)d_in[16];
    float* out = (float*)d_out;

    char* p = (char*)d_ws;
    auto alloc = [&](size_t bytes) -> void* {
        void* r = (void*)p;
        p += (bytes + 255) & ~(size_t)255;
        return r;
    };
    __bf16* xb    = (__bf16*)alloc(8192ull * 1024 * 2);
    __bf16* WqkvT = (__bf16*)alloc(3072ull * 1024 * 2);
    float*  bqkv  = (float*) alloc(3072ull * 4);
    __bf16* WoT   = (__bf16*)alloc(1024ull * 1024 * 2);
    __bf16* W1T   = (__bf16*)alloc(4096ull * 1024 * 2);
    __bf16* W2T   = (__bf16*)alloc(1024ull * 4096 * 2);
    __bf16* qkvb  = (__bf16*)alloc(8192ull * 3072 * 2);
    __bf16* attnb = (__bf16*)alloc(8192ull * 1024 * 2);
    float*  proj  = (float*) alloc(8192ull * 1024 * 4);   // reused for ff2
    float*  inter = (float*) alloc(8192ull * 1024 * 4);
    __bf16* ff1   = (__bf16*)alloc(8192ull * 4096 * 2);
    float*  ff2   = proj;

    cvt_bf16_kernel<<<2048, 256, 0, stream>>>(x, xb, 8192 * 1024);
    pack_qkv_kernel<<<2048, 256, 0, stream>>>(Wq, Wk, Wv, bq, bk, bv, WqkvT, bqkv);
    transp_bf16_kernel<<<1024, 256, 0, stream>>>(Wo, WoT, 1024, 1024);
    transp_bf16_kernel<<<2048, 256, 0, stream>>>(W1, W1T, 4096, 1024);
    transp_bf16_kernel<<<2048, 256, 0, stream>>>(W2, W2T, 1024, 4096);

    // qkv = x @ Wqkv + b  -> bf16 [8192][3072]
    gemm_bf16_kernel<0><<<dim3(24, 64), 256, 0, stream>>>(
        xb, WqkvT, bqkv, (void*)qkvb, 8192, 3072, 1024);
    // flash attention -> bf16 [8192][1024] (heads concatenated)
    attn_kernel<<<dim3(16, 128), 128, 0, stream>>>(qkvb, attnb);
    // proj = attn @ Wo + bo -> f32
    gemm_bf16_kernel<2><<<dim3(8, 64), 256, 0, stream>>>(
        attnb, WoT, bo, (void*)proj, 8192, 1024, 1024);
    // inter = LN1(x + proj)
    ln_add_kernel<<<8192, 256, 0, stream>>>(x, proj, g1, be1, inter);
    // ff1 = relu(x @ W1 + b1) -> bf16 [8192][4096]
    gemm_bf16_kernel<1><<<dim3(32, 64), 256, 0, stream>>>(
        xb, W1T, b1, (void*)ff1, 8192, 4096, 1024);
    // ff2 = ff1 @ W2 + b2 -> f32
    gemm_bf16_kernel<2><<<dim3(8, 64), 256, 0, stream>>>(
        ff1, W2T, b2, (void*)ff2, 8192, 1024, 4096);
    // out = LN2(ff2 + inter)
    ln_add_kernel<<<8192, 256, 0, stream>>>(ff2, inter, g2, be2, out);
}
